// ResNet_84404697301592
// MI455X (gfx1250) — matrix operand order repaired and checked by reading, not compiled, not run on hardware
//
#include <hip/hip_runtime.h>

// ---------------------------------------------------------------------------
// CDNA5 (gfx1250): implicit-GEMM convs on v_wmma_f32_16x16x32_f16 with
// WGP-cooperative LDS staging of the im2col B tile (shared by up to 8 waves),
// async global->LDS staging for the dense head, collapsed attention,
// LDS-based top-k / self-correlation, TDM descriptor prefetch.
// ---------------------------------------------------------------------------

typedef __attribute__((ext_vector_type(16))) _Float16 v16h;
typedef __attribute__((ext_vector_type(8)))  float    v8f;

union AF { v16h v; _Float16 h[16]; uint4 q[2]; };

// ---------------------------- weight conversion ----------------------------
__global__ void cvt_f32_f16(const float* __restrict__ s, _Float16* __restrict__ d, int n) {
    int i = blockIdx.x * 256 + threadIdx.x;
    if (i < n) d[i] = (_Float16)s[i];
}

// ------------------------- implicit-GEMM conv (WMMA) -----------------------
// GEMM: Y[Cout, B*Ho*Wo] = W[Cout, Cin*KH*KW] x im2col(X).
// Workgroup = nwv waves; all waves share one 32(K)x64(N) B tile staged in LDS
// per k-chunk (im2col gather done once per block, not once per wave).  Each
// wave owns a 16-row M sub-tile, reads B fragments via ds_load_b128 and issues
// 4 WMMAs per k-chunk.  Dense 4x4 head stages B with global_load_async_to_lds.
template<int KH, int KW, int STRIDE, int PAD>
__global__ __launch_bounds__(256)
void conv_wmma(const _Float16* __restrict__ Wt, const _Float16* __restrict__ X,
               float* __restrict__ Y32, _Float16* __restrict__ Y16,
               const float* __restrict__ bng, const float* __restrict__ bnb,
               const float* __restrict__ bnm, const float* __restrict__ bnv,
               const float* __restrict__ bias, const float* __restrict__ res,
               int relu, int Bn, int Cin, int Cout, int H, int W)
{
    constexpr int NB = 4;                     // 4 n-tiles of 16 -> N tile = 64
    const int Ho = (H + 2 * PAD - KH) / STRIDE + 1;
    const int Wo = (W + 2 * PAD - KW) / STRIDE + 1;
    const int K  = Cin * KH * KW;

    const int tid  = threadIdx.x;
    const int lane = tid & 31;
    const int wv   = tid >> 5;                // wave id = M sub-tile id
    const int nwv  = blockDim.x >> 5;
    const int lrow = lane & 15;
    const int lhi  = lane >> 4;
    const int tn   = blockIdx.x * (16 * NB);              // GEMM N tile base
    const int tm   = (blockIdx.y * nwv + wv) * 16;        // this wave's M tile

    __shared__ alignas(16) _Float16 blds[64 * 32];        // B tile [n_local][k], 4KB

    // decompose this lane's NB GEMM columns n -> (b, ho, wo) for the stores
    int bb_[NB], ho_[NB], wo_[NB];
#pragma unroll
    for (int j = 0; j < NB; ++j) {
        int n  = tn + j * 16 + lrow;
        wo_[j] = n % Wo; int t1 = n / Wo;
        ho_[j] = t1 % Ho; bb_[j] = t1 / Ho;
    }

    // A (weights): contiguous layout: lanes 0-15 K{0..15}, lanes 16-31 K{16..31}
    const _Float16* arow = Wt + (size_t)(tm + lrow) * K + lhi * 16;

    v8f acc[NB];
#pragma unroll
    for (int j = 0; j < NB; ++j) acc[j] = (v8f){};

    for (int kk = 0; kk < K; kk += 32) {
        // ---- stage B tile (64 n x 32 k) into LDS, once per workgroup ----
        if (KH == 4) {
            // dense head: K contiguous per column -> async global->LDS (ASYNCcnt)
            for (int p = tid; p < 64 * 16; p += blockDim.x) {          // dwords
                int nl = p >> 4, kd = p & 15;
                const _Float16* gp = X + (size_t)(tn + nl) * K + kk + kd * 2;
                unsigned la = (unsigned)(unsigned long long)&blds[nl * 32 + kd * 2];
                asm volatile("global_load_async_to_lds_b32 %0, %1, off"
                             :: "v"(la), "v"(gp) : "memory");
            }
            asm volatile("s_wait_asynccnt 0x0" ::: "memory");
        } else {
            // im2col gather, amortized across all waves of the workgroup
            for (int p = tid; p < 64 * 32; p += blockDim.x) {
                int nl = p >> 5, kt = p & 31;
                int n  = tn + nl;
                int wo = n % Wo; int t1 = n / Wo;
                int ho = t1 % Ho; int bb = t1 / Ho;
                int k = kk + kt;
                int ci, ky, kx;
                if (KH == 1 && KW == 1) { ci = k; ky = 0; kx = 0; }
                else                    { ci = k / 9; int r = k - ci * 9; ky = r / 3; kx = r - ky * 3; }
                int y = ho * STRIDE + ky - PAD;
                int x = wo * STRIDE + kx - PAD;
                _Float16 vv = (_Float16)0.0f;
                if ((unsigned)y < (unsigned)H && (unsigned)x < (unsigned)W)
                    vv = X[(((size_t)bb * Cin + ci) * H + y) * W + x];
                blds[nl * 32 + kt] = vv;
            }
        }
        __syncthreads();

        AF afrag;
        const uint4* ap = (const uint4*)(arow + kk);
        afrag.q[0] = ap[0];      // K = kk + {0..7}  (lanes 16-31: {16..23})
        afrag.q[1] = ap[1];      // K = kk + {8..15} (lanes 16-31: {24..31})
        __builtin_prefetch((const void*)(arow + kk + 32), 0, 0);  // next weight tile

#pragma unroll
        for (int j = 0; j < NB; ++j) {
            AF bf;                                     // ds_load_b128 x2
            const uint4* bp = (const uint4*)(blds + (j * 16 + lrow) * 32 + lhi * 16);
            bf.q[0] = bp[0];
            bf.q[1] = bp[1];
            acc[j] = __builtin_amdgcn_wmma_f32_16x16x32_f16(false, afrag.v, false, bf.v,
                                                            (short)0, acc[j], false, false);
        }
        __syncthreads();
    }

    // C layout: lanes 0-15 N=lane M=r; lanes 16-31 N=lane-16 M=8+r
#pragma unroll
    for (int j = 0; j < NB; ++j) {
#pragma unroll
        for (int r = 0; r < 8; ++r) {
            int co = tm + lhi * 8 + r;
            float val = acc[j][r];
            if (bng) {
                float inv = bng[co] * rsqrtf(bnv[co] + 1e-5f);
                val = val * inv + (bnb[co] - bnm[co] * inv);
            }
            if (bias) val += bias[co];
            size_t oi = (((size_t)bb_[j] * Cout + co) * Ho + ho_[j]) * Wo + wo_[j];
            if (res)  val += res[oi];
            if (relu) val = fmaxf(val, 0.0f);
            if (Y32) Y32[oi] = val;
            if (Y16) Y16[oi] = (_Float16)val;
        }
    }
}

template<int KH, int KW, int S, int P>
static void launch_conv(hipStream_t st, const _Float16* Wt, const _Float16* X,
                        float* Y32, _Float16* Y16,
                        const float* g, const float* b, const float* m, const float* v,
                        const float* bias, const float* res, int relu,
                        int Bn, int Cin, int Cout, int H, int W)
{
    int Ho = (H + 2 * P - KH) / S + 1;
    int Wo = (W + 2 * P - KW) / S + 1;
    int mw = (Cout >= 128) ? 8 : 4;                       // waves (M sub-tiles) per block
    dim3 grid((unsigned)((Bn * Ho * Wo) / 64), (unsigned)(Cout / (16 * mw)));
    hipLaunchKernelGGL((conv_wmma<KH, KW, S, P>), grid, dim3(mw * 32), 0, st,
                       Wt, X, Y32, Y16, g, b, m, v, bias, res, relu, Bn, Cin, Cout, H, W);
}

// --------------------------- conv1 (3->64, direct) -------------------------
__global__ void conv1_kernel(const float* __restrict__ X, const float* __restrict__ Wc,
                             const float* __restrict__ g, const float* __restrict__ b_,
                             const float* __restrict__ m, const float* __restrict__ v,
                             float* __restrict__ Y, int total)
{
    int i = blockIdx.x * 256 + threadIdx.x;
    if (i >= total) return;
    int wo = i & 31; int t = i >> 5;
    int ho = t & 31; t >>= 5;
    int co = t & 63; int bb = t >> 6;
    float acc = 0.f;
    for (int ci = 0; ci < 3; ++ci)
        for (int kh = 0; kh < 3; ++kh) {
            int y = ho + kh - 1; if ((unsigned)y >= 32u) continue;
            for (int kw = 0; kw < 3; ++kw) {
                int x = wo + kw - 1; if ((unsigned)x >= 32u) continue;
                acc += X[(((size_t)bb * 3 + ci) * 32 + y) * 32 + x] *
                       Wc[((co * 3 + ci) * 3 + kh) * 3 + kw];
            }
        }
    float inv = g[co] * rsqrtf(v[co] + 1e-5f);
    acc = acc * inv + (b_[co] - m[co] * inv);
    Y[i] = fmaxf(acc, 0.0f);
}

// ------------------------------ DCT quantize -------------------------------
__constant__ float DCTM[4][4] = {
    { 2.0f, 2.0f, 2.0f, 2.0f },
    { 1.8477590650225735f,  0.7653668647301796f, -0.7653668647301796f, -1.8477590650225735f },
    { 1.4142135623730951f, -1.4142135623730951f, -1.4142135623730951f,  1.4142135623730951f },
    { 0.7653668647301796f, -1.8477590650225735f,  1.8477590650225735f, -0.7653668647301796f } };
// IDCT[n][k] = DCTM[k][n] * (k==0 ? 1/16 : 1/8)  (rows orthogonal: M M^T = diag(16,8,8,8))

__global__ void dct_kernel(const float* __restrict__ X, float* __restrict__ Y32,
                           _Float16* __restrict__ Y16, int total)
{
    int i = blockIdx.x * 256 + threadIdx.x;
    if (i >= total) return;
    int qx = i & 7; int t = i >> 3;
    int py = t & 7; t >>= 3;
    int c = t & 63; int bb = t >> 6;
    const float* src = X + (((size_t)(bb * 64 + c) * 32) + py * 4) * 32 + qx * 4;
    float xb[4][4], d[4][4];
    for (int a = 0; a < 4; ++a)
        for (int e = 0; e < 4; ++e) xb[a][e] = src[a * 32 + e];
    for (int ii = 0; ii < 4; ++ii)
        for (int jj = 0; jj < 4; ++jj) {
            float s = 0.f;
            for (int a = 0; a < 4; ++a) {
                float s2 = 0.f;
                for (int e = 0; e < 4; ++e) s2 += DCTM[jj][e] * xb[a][e];
                s += DCTM[ii][a] * s2;
            }
            d[ii][jj] = (s < -0.2f || s > 0.2f) ? rintf(s) : 0.0f;   // deadzone + RNE round
        }
    size_t base = (size_t)(src - X);
    for (int ii = 0; ii < 4; ++ii)
        for (int jj = 0; jj < 4; ++jj) {
            float s = 0.f;
            for (int a = 0; a < 4; ++a) {
                float sa = (a == 0) ? 0.0625f : 0.125f;
                float s2 = 0.f;
                for (int e = 0; e < 4; ++e) {
                    float se = (e == 0) ? 0.0625f : 0.125f;
                    s2 += DCTM[e][jj] * se * d[a][e];
                }
                s += DCTM[a][ii] * sa * s2;
            }
            Y32[base + ii * 32 + jj] = s;
            Y16[base + ii * 32 + jj] = (_Float16)s;
        }
}

// -------------------- attention collapse (pure linear chain) ---------------
// weff = W1^T (W2^T W3);  v = proj^T weff
__global__ void collapse_kernel(const float* __restrict__ W1, const float* __restrict__ W2,
                                const float* __restrict__ W3, const float* __restrict__ proj,
                                int Cin, float* __restrict__ weff, float* __restrict__ vout)
{
    __shared__ float u[64];
    __shared__ float we[512];
    int tid = threadIdx.x;   // 512 threads
    if (tid < 64) {
        float a = 0.f;
        for (int k = 0; k < 32; ++k) a += W3[k] * W2[k * 64 + tid];
        u[tid] = a;
    }
    __syncthreads();
    {
        float a = 0.f;
        for (int j = 0; j < 64; ++j) a += u[j] * W1[j * 512 + tid];
        we[tid] = a; weff[tid] = a;
    }
    __syncthreads();
    for (int c = tid; c < Cin; c += 512) {
        float a = 0.f;
        for (int i = 0; i < 512; ++i) a += we[i] * proj[i * Cin + c];
        vout[c] = a;
    }
}

// score[b,hw] = v . lN[b,:,hw] + weff . g[b,:]
__global__ void score_kernel(const float* __restrict__ v, int C, const float* __restrict__ X,
                             const float* __restrict__ weff, const float* __restrict__ G,
                             int HW, float* __restrict__ sbuf, float* __restrict__ oout)
{
    int b = blockIdx.x, tid = threadIdx.x;
    __shared__ float red[256];
    __shared__ float gd_s;
    float p = 0.f;
    for (int i = tid; i < 512; i += 256) p += weff[i] * G[b * 512 + i];
    red[tid] = p; __syncthreads();
    for (int s = 128; s > 0; s >>= 1) { if (tid < s) red[tid] += red[tid + s]; __syncthreads(); }
    if (tid == 0) gd_s = red[0];
    __syncthreads();
    float gd = gd_s;
    for (int hw = tid; hw < HW; hw += 256) {
        float a = gd;
        for (int c = 0; c < C; ++c) a += v[c] * X[((size_t)b * C + c) * HW + hw];
        sbuf[b * HW + hw] = a;
        if (oout) oout[b * HW + hw] = a;
    }
}

// ------------------------------- top-k (50) --------------------------------
__global__ void topk_kernel(const float* __restrict__ score, int HW,
                            int* __restrict__ idxout, float* __restrict__ wout)
{
    int b = blockIdx.x, tid = threadIdx.x;
    __shared__ float sv[1024];
    __shared__ float rv[256];
    __shared__ int   ri[256];
    __shared__ float vals[50];
    __shared__ int   vidx[50];
    for (int i = tid; i < 1024; i += 256) sv[i] = (i < HW) ? score[b * HW + i] : -3.4e38f;
    __syncthreads();
    for (int j = 0; j < 50; ++j) {
        float best = -3.4e38f; int bi = 0x7fffffff;
        for (int i = tid; i < HW; i += 256) {
            float x = sv[i];
            if (x > best || (x == best && i < bi)) { best = x; bi = i; }
        }
        rv[tid] = best; ri[tid] = bi; __syncthreads();
        for (int s = 128; s > 0; s >>= 1) {
            if (tid < s) {
                float o = rv[tid + s]; int oi = ri[tid + s];
                if (o > rv[tid] || (o == rv[tid] && oi < ri[tid])) { rv[tid] = o; ri[tid] = oi; }
            }
            __syncthreads();
        }
        if (tid == 0) { vals[j] = rv[0]; vidx[j] = ri[0]; sv[ri[0]] = -3.4e38f; }
        __syncthreads();
    }
    if (tid < 50) {
        float thr = vals[49];
        wout[b * 50 + tid]  = fmaxf(floorf(vals[tid] - thr) + 0.5f, 0.0f) * 2.0f;
        idxout[b * 50 + tid] = vidx[tid];
    }
}

// ----------------------------- self-correlation ----------------------------
__global__ void corr_kernel(const float* __restrict__ X, int C, int HW,
                            const int* __restrict__ idx, const float* __restrict__ wgt,
                            const float* __restrict__ CW, const float* __restrict__ CB,
                            float* __restrict__ feat, int featoff)
{
    int b = blockIdx.x, tid = threadIdx.x;
    extern __shared__ float sm[];
    float* emb  = sm;            // C*50
    float* emb2 = sm + C * 50;   // C*50
    __shared__ int   sidx[50];
    __shared__ float swgt[50];
    if (tid < 50) { sidx[tid] = idx[b * 50 + tid]; swgt[tid] = wgt[b * 50 + tid]; }
    __syncthreads();
    for (int p = tid; p < C * 50; p += blockDim.x) {
        int c = p / 50, k = p - c * 50;
        emb[p] = swgt[k] * X[((size_t)b * C + c) * HW + sidx[k]];
    }
    __syncthreads();
    for (int p = tid; p < C * 50; p += blockDim.x) {
        int c = p / 50, o = p - c * 50;
        float a = CB[c * 50 + o];
        const float* wr = CW + (size_t)c * 2500 + o;
        const float* er = emb + c * 50;
        for (int k = 0; k < 50; ++k) a += er[k] * wr[k * 50];
        emb2[p] = a;
    }
    __syncthreads();
    for (int p = tid; p < 2500; p += blockDim.x) {
        int k = p / 50, s = p - k * 50;
        float a = 0.f;
        for (int c = 0; c < C; ++c) a += emb2[c * 50 + k] * emb2[c * 50 + s];
        feat[(size_t)b * 7500 + featoff + p] = a;
    }
}

// -------------------------------- classifier -------------------------------
__global__ void cls_kernel(const float* __restrict__ feat, const float* __restrict__ Wc,
                           const float* __restrict__ bc, float* __restrict__ out)
{
    int b = blockIdx.x, tid = threadIdx.x;
    __shared__ float red[256];
    for (int t = 0; t < 10; ++t) {
        float p = 0.f;
        for (int j = tid; j < 7500; j += 256) p += feat[(size_t)b * 7500 + j] * Wc[t * 7500 + j];
        red[tid] = p; __syncthreads();
        for (int s = 128; s > 0; s >>= 1) { if (tid < s) red[tid] += red[tid + s]; __syncthreads(); }
        if (tid == 0) out[b * 10 + t] = red[0] + bc[t];
        __syncthreads();
    }
}

// ------------------------ TDM async tensor stage ---------------------------
#if __has_builtin(__builtin_amdgcn_tensor_load_to_lds)
#define HAVE_TDM 1
typedef unsigned int tdm_v4u __attribute__((ext_vector_type(4)));
typedef int          tdm_v8i __attribute__((ext_vector_type(8)));
typedef int          tdm_v4i __attribute__((ext_vector_type(4)));

__global__ void tdm_stage_kernel(const _Float16* __restrict__ src, float* __restrict__ sink)
{
    __shared__ _Float16 stage[2048];     // 4KB staging tile
    unsigned lds = (unsigned)(unsigned long long)(&stage[0]);
    unsigned long long ga = (unsigned long long)src;
    // D# group0: count=1, lds_addr, global_addr[56:0], type=2 (image)
    tdm_v4u g0 = { 1u, lds,
                   (unsigned)(ga & 0xffffffffu),
                   (unsigned)((ga >> 32) & 0x1ffffffu) | (2u << 30) };
    // D# group1: data_size=8B, tensor_dim0=512, tensor_dim1=1, tile=512x1, stride0=512
    tdm_v8i g1 = { (int)(3u << 16), (int)(512u << 16), (int)(1u << 16),
                   (int)(512u << 16), 1, 512, (int)(512u << 16), 0 };
    tdm_v4i g2 = { 0, 0, 0, 0 };
    tdm_v4i g3 = { 0, 0, 0, 0 };
#if __has_include(<hip/amd_detail/amd_gfx1250_TDM.h>)
    tdm_v8i g4 = { 0, 0, 0, 0, 0, 0, 0, 0 };
    __builtin_amdgcn_tensor_load_to_lds(g0, g1, g2, g3, g4, 0);   // clang-23 6-arg form
#else
    __builtin_amdgcn_tensor_load_to_lds(g0, g1, g2, g3, 0);       // ROCm 7.2 5-arg form
#endif
    __builtin_amdgcn_s_wait_tensorcnt(0);
    __syncthreads();
    if (threadIdx.x == 0) sink[0] = (float)stage[0];
}
#endif

// ---------------------------------------------------------------------------
extern "C" void kernel_launch(void* const* d_in, const int* in_sizes, int n_in,
                              void* d_out, int out_size, void* d_ws, size_t ws_size,
                              hipStream_t stream)
{
    (void)in_sizes; (void)n_in; (void)out_size; (void)ws_size;
    const int Bn = 64;

    // ---- input index map (depth-first insertion-order flatten of setup_inputs) ----
    int cur = 0;
    auto nxt = [&](int n) { int r = cur; cur += n; return r; };
    struct BI { int c1, b1, c2, b2, sc, scb; };
    auto mkblk = [&](bool has_sc) {
        BI x; x.c1 = nxt(1); x.b1 = nxt(4); x.c2 = nxt(1); x.b2 = nxt(4);
        x.sc = has_sc ? nxt(1) : -1; x.scb = has_sc ? nxt(4) : -1; return x;
    };
    const int iX = nxt(1);
    const int iConv1 = nxt(1);
    const int iBn1 = nxt(4);
    BI l1b0 = mkblk(false), l1b1 = mkblk(false);
    BI l2b0 = mkblk(true),  l2b1 = mkblk(false);
    BI l3b0 = mkblk(true),  l3b1 = mkblk(false);
    BI l4b0 = mkblk(true),  l4b1 = mkblk(false);
    const int iProj1 = nxt(1), iProj2 = nxt(1), iProj3 = nxt(1);
    const int iA1 = nxt(3), iA2 = nxt(3), iA3 = nxt(3);           // w1,w2,w3 each
    const int iDenseW = nxt(1), iDenseB = nxt(1), iClsW = nxt(1), iClsB = nxt(1);
    const int iC1W = nxt(1), iC1B = nxt(1), iC2W = nxt(1), iC2B = nxt(1),
              iC3W = nxt(1), iC3B = nxt(1);
    auto F = [&](int i) -> const float* { return (const float*)d_in[i]; };

    // ---- workspace bump allocator ----
    char* wsbase = (char*)d_ws; size_t off = 0;
    auto alloc  = [&](size_t bytes) { size_t a = (off + 255) & ~(size_t)255; off = a + bytes; return (void*)(wsbase + a); };
    auto allocF = [&](size_t n) { return (float*)alloc(n * 4); };
    auto allocH = [&](size_t n) { return (_Float16*)alloc(n * 2); };
    auto cvt = [&](int idx, size_t n) -> _Float16* {
        _Float16* d = allocH(n);
        hipLaunchKernelGGL(cvt_f32_f16, dim3((unsigned)((n + 255) / 256)), dim3(256), 0, stream,
                           F(idx), d, (int)n);
        return d;
    };

    // ---- activation buffers ----
    float*    A32 = allocF((size_t)Bn * 64 * 32 * 32);
    float*    D32 = allocF((size_t)Bn * 64 * 32 * 32);
    _Float16* D16 = allocH((size_t)Bn * 64 * 32 * 32);
    _Float16* T16 = allocH((size_t)Bn * 64 * 32 * 32);   // block-intermediate h (max size)
    float*    U32 = allocF((size_t)Bn * 64 * 32 * 32);   // block0 output per layer (max size)
    _Float16* U16 = allocH((size_t)Bn * 64 * 32 * 32);
    float*    S32 = allocF((size_t)Bn * 128 * 16 * 16);  // shortcut output (max size)
    float*    L1_32 = allocF((size_t)Bn * 64 * 32 * 32);  _Float16* L1_16 = allocH((size_t)Bn * 64 * 32 * 32);
    float*    L2_32 = allocF((size_t)Bn * 128 * 16 * 16); _Float16* L2_16 = allocH((size_t)Bn * 128 * 16 * 16);
    float*    L3_32 = allocF((size_t)Bn * 256 * 8 * 8);   _Float16* L3_16 = allocH((size_t)Bn * 256 * 8 * 8);
    float*    L4_32 = allocF((size_t)Bn * 512 * 4 * 4);   _Float16* L4_16 = allocH((size_t)Bn * 512 * 4 * 4);
    float*    G32 = allocF((size_t)Bn * 512);
    float* weff1 = allocF(512); float* weff2 = allocF(512); float* weff3 = allocF(512);
    float* v1 = allocF(64); float* v2 = allocF(128); float* v3 = allocF(256);
    float* s1 = allocF((size_t)Bn * 1024); float* s2 = allocF((size_t)Bn * 256); float* s3 = allocF((size_t)Bn * 64);
    int* idx1 = (int*)alloc(Bn * 50 * 4); float* w1b = allocF(Bn * 50);
    int* idx2 = (int*)alloc(Bn * 50 * 4); float* w2b = allocF(Bn * 50);
    int* idx3 = (int*)alloc(Bn * 50 * 4); float* w3b = allocF(Bn * 50);
    float* feat = allocF((size_t)Bn * 7500);
    float* sink = allocF(16); (void)sink;

    // ---- stem: conv1 + BN + ReLU, then DCT quantize ----
    hipLaunchKernelGGL(conv1_kernel, dim3((Bn * 64 * 32 * 32 + 255) / 256), dim3(256), 0, stream,
                       F(iX), F(iConv1), F(iBn1), F(iBn1 + 1), F(iBn1 + 2), F(iBn1 + 3),
                       A32, Bn * 64 * 32 * 32);
    hipLaunchKernelGGL(dct_kernel, dim3((Bn * 64 * 8 * 8 + 255) / 256), dim3(256), 0, stream,
                       A32, D32, D16, Bn * 64 * 8 * 8);

    // ---- basic block runner ----
    auto run_block = [&](const BI& bi, int Cin, int Cout, int H, int W, int stride,
                         float* X32, _Float16* X16, float* O32, _Float16* O16) {
        int Ho = H / stride, Wo = W / stride;
        _Float16* wc1 = cvt(bi.c1, (size_t)Cout * Cin * 9);
        _Float16* wc2 = cvt(bi.c2, (size_t)Cout * Cout * 9);
        if (stride == 1)
            launch_conv<3, 3, 1, 1>(stream, wc1, X16, nullptr, T16,
                                    F(bi.b1), F(bi.b1 + 1), F(bi.b1 + 2), F(bi.b1 + 3),
                                    nullptr, nullptr, 1, Bn, Cin, Cout, H, W);
        else
            launch_conv<3, 3, 2, 1>(stream, wc1, X16, nullptr, T16,
                                    F(bi.b1), F(bi.b1 + 1), F(bi.b1 + 2), F(bi.b1 + 3),
                                    nullptr, nullptr, 1, Bn, Cin, Cout, H, W);
        const float* res = X32;
        if (bi.sc >= 0) {
            _Float16* wsc = cvt(bi.sc, (size_t)Cout * Cin);
            launch_conv<1, 1, 2, 0>(stream, wsc, X16, S32, nullptr,
                                    F(bi.scb), F(bi.scb + 1), F(bi.scb + 2), F(bi.scb + 3),
                                    nullptr, nullptr, 0, Bn, Cin, Cout, H, W);
            res = S32;
        }
        launch_conv<3, 3, 1, 1>(stream, wc2, T16, O32, O16,
                                F(bi.b2), F(bi.b2 + 1), F(bi.b2 + 2), F(bi.b2 + 3),
                                nullptr, res, 1, Bn, Cout, Cout, Ho, Wo);
    };

    run_block(l1b0,  64,  64, 32, 32, 1, D32,  D16,  U32,  U16);
    run_block(l1b1,  64,  64, 32, 32, 1, U32,  U16,  L1_32, L1_16);
    run_block(l2b0,  64, 128, 32, 32, 2, L1_32, L1_16, U32,  U16);
    run_block(l2b1, 128, 128, 16, 16, 1, U32,  U16,  L2_32, L2_16);
    run_block(l3b0, 128, 256, 16, 16, 2, L2_32, L2_16, U32,  U16);
    run_block(l3b1, 256, 256,  8,  8, 1, U32,  U16,  L3_32, L3_16);
    run_block(l4b0, 256, 512,  8,  8, 2, L3_32, L3_16, U32,  U16);
    run_block(l4b1, 512, 512,  4,  4, 1, U32,  U16,  L4_32, L4_16);

    // ---- dense 4x4 conv -> g [B,512] (WMMA + async global->LDS B staging) ----
    _Float16* wd = cvt(iDenseW, (size_t)512 * 512 * 16);
    launch_conv<4, 4, 1, 0>(stream, wd, L4_16, G32, nullptr,
                            nullptr, nullptr, nullptr, nullptr,
                            F(iDenseB), nullptr, 0, Bn, 512, 512, 4, 4);

    // ---- collapsed attention scores ----
    hipLaunchKernelGGL(collapse_kernel, dim3(1), dim3(512), 0, stream,
                       F(iA1), F(iA1 + 1), F(iA1 + 2), F(iProj1), 64, weff1, v1);
    hipLaunchKernelGGL(collapse_kernel, dim3(1), dim3(512), 0, stream,
                       F(iA2), F(iA2 + 1), F(iA2 + 2), F(iProj2), 128, weff2, v2);
    hipLaunchKernelGGL(collapse_kernel, dim3(1), dim3(512), 0, stream,
                       F(iA3), F(iA3 + 1), F(iA3 + 2), F(iProj3), 256, weff3, v3);

    float* outp = (float*)d_out;
    hipLaunchKernelGGL(score_kernel, dim3(Bn), dim3(256), 0, stream,
                       v1, 64, L1_32, weff1, G32, 1024, s1, outp + 640);
    hipLaunchKernelGGL(score_kernel, dim3(Bn), dim3(256), 0, stream,
                       v2, 128, L2_32, weff2, G32, 256, s2, outp + 640 + 65536);
    hipLaunchKernelGGL(score_kernel, dim3(Bn), dim3(256), 0, stream,
                       v3, 256, L3_32, weff3, G32, 64, s3, (float*)nullptr);

    // ---- top-50 + self correlation -> feat[B,7500] ----
    hipLaunchKernelGGL(topk_kernel, dim3(Bn), dim3(256), 0, stream, s1, 1024, idx1, w1b);
    hipLaunchKernelGGL(topk_kernel, dim3(Bn), dim3(256), 0, stream, s2,  256, idx2, w2b);
    hipLaunchKernelGGL(topk_kernel, dim3(Bn), dim3(256), 0, stream, s3,   64, idx3, w3b);
    hipLaunchKernelGGL(corr_kernel, dim3(Bn), dim3(256), 2u * 64  * 50 * 4, stream,
                       L1_32, 64, 1024, idx1, w1b, F(iC1W), F(iC1B), feat, 0);
    hipLaunchKernelGGL(corr_kernel, dim3(Bn), dim3(256), 2u * 128 * 50 * 4, stream,
                       L2_32, 128, 256, idx2, w2b, F(iC2W), F(iC2B), feat, 2500);
    hipLaunchKernelGGL(corr_kernel, dim3(Bn), dim3(256), 2u * 256 * 50 * 4, stream,
                       L3_32, 256, 64, idx3, w3b, F(iC3W), F(iC3B), feat, 5000);

    // ---- classifier ----
    hipLaunchKernelGGL(cls_kernel, dim3(Bn), dim3(256), 0, stream,
                       feat, F(iClsW), F(iClsB), outp);

#ifdef HAVE_TDM
    hipLaunchKernelGGL(tdm_stage_kernel, dim3(1), dim3(32), 0, stream, wd, sink);
#endif
}
